// GATLayerImp3_56289841381933
// MI455X (gfx1250) — matrix operand order repaired and checked by reading, not compiled, not run on hardware
//
#include <hip/hip_runtime.h>
#include <hip/hip_bf16.h>
#include <math.h>

// ---------------------------------------------------------------------------
// GAT layer for MI455X (gfx1250, wave32, WMMA, async-to-LDS).
// Fused GEMM  P = Xbf16 @ [W | skip_W]bf16  via v_wmma_f32_16x16x32_bf16,
// A-tile staged with GLOBAL_LOAD_ASYNC_TO_LDS_B128 (ASYNCcnt path),
// then edge softmax-aggregation with coalesced atomics, skip+bias+ELU.
// ---------------------------------------------------------------------------

typedef __attribute__((ext_vector_type(16))) __bf16        v16bf;
typedef __attribute__((ext_vector_type(8)))  float         v8f;
typedef __attribute__((ext_vector_type(4)))  unsigned int  uint4v;

#define FIN   256
#define HEADS 8
#define FEAT  32
#define HF    256           // HEADS*FEAT
#define NCOLS 512           // [W | skip_W]
#define LDA   264           // padded bf16 LDS row stride (528B -> 132 dwords -> conflict-free)

__device__ __forceinline__ unsigned short f2bf(float f) {
    unsigned int u = __float_as_uint(f);
    u += 0x7fffu + ((u >> 16) & 1u);          // round-to-nearest-even
    return (unsigned short)(u >> 16);
}

__device__ __forceinline__ unsigned int encodeOrderedF32(float f) {
    unsigned int b = __float_as_uint(f);
    return (b & 0x80000000u) ? ~b : (b | 0x80000000u);
}
__device__ __forceinline__ float decodeOrderedF32(unsigned int k) {
    unsigned int b = (k & 0x80000000u) ? (k & 0x7fffffffu) : ~k;
    return __uint_as_float(b);
}

// ---------------------------------------------------------------------------
// 1) convert x (f32) -> bf16 row-major
__global__ void convert_x_kernel(const float* __restrict__ x,
                                 unsigned short* __restrict__ xbf, int total) {
    int i = blockIdx.x * blockDim.x + threadIdx.x;
    if (i < total) xbf[i] = f2bf(x[i]);
}

// 2) pack [W | skip_W] -> Bc bf16, COLUMN-major: Bc[n*256 + k]
__global__ void pack_w_kernel(const float* __restrict__ W,
                              const float* __restrict__ skip_W,
                              unsigned short* __restrict__ Bc) {
    int i = blockIdx.x * blockDim.x + threadIdx.x;   // n*256 + k
    if (i >= NCOLS * FIN) return;
    int n = i >> 8, k = i & 255;
    float v = (n < HF) ? W[(size_t)k * HF + n] : skip_W[(size_t)k * HF + (n - HF)];
    Bc[i] = f2bf(v);
}

// ---------------------------------------------------------------------------
// 3) WMMA GEMM: P(M x 512) = Xbf(M x 256) @ Bc(256 x 512)
//    cols 0..255 -> proj ; cols 256..511 -> out = skip + bias  (init for agg)
__global__ __launch_bounds__(256)
void gat_gemm_wmma_kernel(const unsigned short* __restrict__ Xbf,
                          const unsigned short* __restrict__ Bc,
                          const float* __restrict__ bias,
                          float* __restrict__ proj,
                          float* __restrict__ out,
                          int Nn) {
    __shared__ unsigned short sA[64 * LDA];          // 33,792 B

    const int tid   = threadIdx.x;
    const int wave  = tid >> 5;
    const int lane  = tid & 31;
    const int half  = lane >> 4;
    const int l16   = lane & 15;
    const int waveM = wave & 3;                      // 4 waves along M
    const int waveN = wave >> 2;                     // 2 waves along N

    const int mBlock = blockIdx.x * 64;
    const int nBlock = blockIdx.y * 64;

    // ---- stage A tile: 64 rows x 256 K (bf16) via async memory->LDS -------
    // Each lane moves one 16B chunk per iteration, no VGPR staging (ASYNCcnt).
    #pragma unroll
    for (int it = 0; it < 8; ++it) {
        int id = tid + it * 256;                     // 0..2047 chunks of 16B
        int r  = id >> 5;                            // row 0..63
        int c  = id & 31;                            // 16B chunk in row
        int gr = mBlock + r; if (gr > Nn - 1) gr = Nn - 1;
        unsigned long long gaddr =
            (unsigned long long)(Xbf + (size_t)gr * FIN + c * 8);
        unsigned ldsa =
            (unsigned)(unsigned long long)(sA + r * LDA + c * 8); // low 32b = LDS offset
        asm volatile("global_load_async_to_lds_b128 %0, %1, off"
                     :: "v"(ldsa), "v"(gaddr) : "memory");
    }
    asm volatile("s_wait_asynccnt 0" ::: "memory");
    __syncthreads();

    // ---- main K loop: 8 steps of 32, 2 WMMAs per step ----------------------
    v8f acc0 = {0.f,0.f,0.f,0.f,0.f,0.f,0.f,0.f};
    v8f acc1 = {0.f,0.f,0.f,0.f,0.f,0.f,0.f,0.f};

    const unsigned short* aBase  = sA + (waveM * 16 + l16) * LDA;
    const int n0 = nBlock + waveN * 32 + l16;
    const unsigned short* b0Base = Bc + (size_t)n0 * FIN;
    const unsigned short* b1Base = Bc + (size_t)(n0 + 16) * FIN;

    #pragma unroll
    for (int kk = 0; kk < FIN; kk += 32) {
        union { v16bf v; uint4v q[2]; } af, bf0, bf1;
        // A fragment: 16 contiguous K at kk+16h (matches B)
        af.q[0] = *(const uint4v*)(aBase + kk + 16 * half);
        af.q[1] = *(const uint4v*)(aBase + kk + 16 * half + 8);
        // B fragments: 16 contiguous K at kk+16h (column-major Bc)
        const unsigned short* bp0 = b0Base + kk + 16 * half;
        const unsigned short* bp1 = b1Base + kk + 16 * half;
        bf0.q[0] = *(const uint4v*)(bp0);
        bf0.q[1] = *(const uint4v*)(bp0 + 8);
        bf1.q[0] = *(const uint4v*)(bp1);
        bf1.q[1] = *(const uint4v*)(bp1 + 8);

        acc0 = __builtin_amdgcn_wmma_f32_16x16x32_bf16(false, af.v, false, bf0.v,
                                                       (short)0, acc0, false, false);
        acc1 = __builtin_amdgcn_wmma_f32_16x16x32_bf16(false, af.v, false, bf1.v,
                                                       (short)0, acc1, false, false);
    }

    // ---- epilogue ----------------------------------------------------------
    v8f accs[2] = {acc0, acc1};
    #pragma unroll
    for (int t = 0; t < 2; ++t) {
        int nc = nBlock + waveN * 32 + t * 16 + l16;
        #pragma unroll
        for (int v = 0; v < 8; ++v) {
            int gr = mBlock + waveM * 16 + v + 8 * half;
            if (gr < Nn) {
                float val = accs[t][v];
                if (nc < HF) proj[(size_t)gr * HF + nc] = val;
                else         out[(size_t)gr * HF + (nc - HF)] = val + bias[nc - HF];
            }
        }
    }
}

// ---------------------------------------------------------------------------
// 4) per-node attention logits (N,H)
__global__ void scores_kernel(const float* __restrict__ proj,
                              const float* __restrict__ a_src,
                              const float* __restrict__ a_trg,
                              float* __restrict__ ssrc,
                              float* __restrict__ strg, int Nn) {
    int i = blockIdx.x * blockDim.x + threadIdx.x;   // n*8 + h
    if (i >= Nn * HEADS) return;
    int n = i >> 3, h = i & 7;
    const float* p  = proj + (size_t)n * HF + h * FEAT;
    const float* as = a_src + h * FEAT;
    const float* at = a_trg + h * FEAT;
    float s1 = 0.f, s2 = 0.f;
    #pragma unroll
    for (int f = 0; f < FEAT; ++f) { float pv = p[f]; s1 += pv * as[f]; s2 += pv * at[f]; }
    ssrc[i] = s1; strg[i] = s2;
}

// 5) per-call re-init: zero denom, reset global-max accumulator
__global__ void init_kernel(float* __restrict__ denom, unsigned int* __restrict__ max_enc, int total) {
    int i = blockIdx.x * blockDim.x + threadIdx.x;
    if (i < total) denom[i] = 0.f;
    if (i == 0) *max_enc = encodeOrderedF32(-3.0e38f);
}

// 6) edge leaky-relu scores + global max (ordered-uint atomic max)
__global__ void edge_score_kernel(const long long* __restrict__ ei,
                                  const float* __restrict__ ssrc,
                                  const float* __restrict__ strg,
                                  float* __restrict__ scores,
                                  unsigned int* __restrict__ max_enc, int Ee) {
    __shared__ float smax[256];
    int i = blockIdx.x * blockDim.x + threadIdx.x;   // e*8 + h
    float s = -3.0e38f;
    if (i < Ee * HEADS) {
        int e = i >> 3, h = i & 7;
        long long src = ei[e], trg = ei[Ee + e];
        float v = ssrc[src * HEADS + h] + strg[trg * HEADS + h];
        v = (v > 0.f) ? v : 0.2f * v;                // leaky_relu 0.2
        scores[i] = v;
        s = v;
    }
    smax[threadIdx.x] = s; __syncthreads();
    for (int off = 128; off > 0; off >>= 1) {
        if ((int)threadIdx.x < off) smax[threadIdx.x] = fmaxf(smax[threadIdx.x], smax[threadIdx.x + off]);
        __syncthreads();
    }
    if (threadIdx.x == 0) atomicMax(max_enc, encodeOrderedF32(smax[0]));
}

// 7) exp(score - max) + segment-sum denom over target
__global__ void edge_exp_kernel(const long long* __restrict__ ei,
                                float* __restrict__ scores,
                                float* __restrict__ denom,
                                const unsigned int* __restrict__ max_enc, int Ee) {
    int i = blockIdx.x * blockDim.x + threadIdx.x;   // e*8 + h
    if (i >= Ee * HEADS) return;
    int e = i >> 3, h = i & 7;
    float m = decodeOrderedF32(*max_enc);
    float v = __expf(scores[i] - m);
    scores[i] = v;
    long long trg = ei[Ee + e];
    atomicAdd(&denom[trg * HEADS + h], v);
}

// 8) weighted scatter aggregation: one wave per edge, fully coalesced per head
__global__ void aggregate_kernel(const long long* __restrict__ ei,
                                 const float* __restrict__ proj,
                                 const float* __restrict__ scores,
                                 const float* __restrict__ denom,
                                 float* __restrict__ out, int Ee) {
    int gid  = blockIdx.x * blockDim.x + threadIdx.x;
    int edge = gid >> 5;
    int lane = gid & 31;
    if (edge >= Ee) return;
    long long src = ei[edge], trg = ei[Ee + edge];
    const float* p = proj + (size_t)src * HF;
    float*       o = out  + (size_t)trg * HF;
    #pragma unroll
    for (int h = 0; h < HEADS; ++h) {
        float att = scores[(size_t)edge * HEADS + h] /
                    (denom[trg * HEADS + h] + 1e-16f);
        int c = h * FEAT + lane;                     // 32 contiguous floats / wave
        atomicAdd(o + c, p[c] * att);
    }
}

// 9) in-place ELU (out already holds agg + skip + bias)
__global__ void elu_kernel(float* __restrict__ out, int total) {
    int i = blockIdx.x * blockDim.x + threadIdx.x;
    if (i >= total) return;
    float v = out[i];
    out[i] = (v > 0.f) ? v : expm1f(v);
}

// ---------------------------------------------------------------------------
extern "C" void kernel_launch(void* const* d_in, const int* in_sizes, int n_in,
                              void* d_out, int out_size, void* d_ws, size_t ws_size,
                              hipStream_t stream) {
    const float*     x      = (const float*)d_in[0];
    const long long* ei     = (const long long*)d_in[1];
    const float*     W      = (const float*)d_in[2];
    const float*     a_src  = (const float*)d_in[3];
    const float*     a_trg  = (const float*)d_in[4];
    const float*     skip_W = (const float*)d_in[5];
    const float*     bias   = (const float*)d_in[6];
    float*           out    = (float*)d_out;

    const int Nn = in_sizes[0] / FIN;       // 50000
    const int Ee = in_sizes[1] / 2;         // 800000

    // ---- workspace layout (256B aligned) ----------------------------------
    char* ws = (char*)d_ws;
    size_t off = 0;
    auto take = [&](size_t bytes) {
        size_t cur = off;
        off = (off + bytes + 255) & ~(size_t)255;
        return ws + cur;
    };
    unsigned short* Xbf     = (unsigned short*)take((size_t)Nn * FIN * 2);
    unsigned short* Bc      = (unsigned short*)take((size_t)NCOLS * FIN * 2);
    float*          proj    = (float*)take((size_t)Nn * HF * 4);
    float*          ssrc    = (float*)take((size_t)Nn * HEADS * 4);
    float*          strg    = (float*)take((size_t)Nn * HEADS * 4);
    float*          scores  = (float*)take((size_t)Ee * HEADS * 4);
    float*          denom   = (float*)take((size_t)Nn * HEADS * 4);
    unsigned int*   max_enc = (unsigned int*)take(256);

    const int TB = 256;

    // 1) x -> bf16
    {
        int total = Nn * FIN;
        convert_x_kernel<<<(total + TB - 1) / TB, TB, 0, stream>>>(x, Xbf, total);
    }
    // 2) pack [W | skip_W] column-major bf16
    pack_w_kernel<<<(NCOLS * FIN + TB - 1) / TB, TB, 0, stream>>>(W, skip_W, Bc);

    // 3) WMMA GEMM -> proj + out-init(skip+bias)
    {
        dim3 grid((Nn + 63) / 64, NCOLS / 64);      // 782 x 8
        gat_gemm_wmma_kernel<<<grid, TB, 0, stream>>>(Xbf, Bc, bias, proj, out, Nn);
    }
    // 4) per-node logits
    scores_kernel<<<(Nn * HEADS + TB - 1) / TB, TB, 0, stream>>>(proj, a_src, a_trg, ssrc, strg, Nn);
    // 5) init denom + max
    init_kernel<<<(Nn * HEADS + TB - 1) / TB, TB, 0, stream>>>(denom, max_enc, Nn * HEADS);
    // 6) edge scores + global max
    edge_score_kernel<<<(Ee * HEADS + TB - 1) / TB, TB, 0, stream>>>(ei, ssrc, strg, scores, max_enc, Ee);
    // 7) exp + denom segment sum
    edge_exp_kernel<<<(Ee * HEADS + TB - 1) / TB, TB, 0, stream>>>(ei, scores, denom, max_enc, Ee);
    // 8) weighted scatter aggregation (one wave per edge, coalesced)
    {
        long long threads = (long long)Ee * 32;
        aggregate_kernel<<<(unsigned)((threads + TB - 1) / TB), TB, 0, stream>>>(ei, proj, scores, denom, out, Ee);
    }
    // 9) ELU
    {
        int total = Nn * HF;
        elu_kernel<<<(total + TB - 1) / TB, TB, 0, stream>>>(out, total);
    }
}
